// MultiHeadAttention_61924838473825
// MI455X (gfx1250) — matrix operand order repaired and checked by reading, not compiled, not run on hardware
//
#include <hip/hip_runtime.h>
#include <hip/hip_bf16.h>

// MHA block for B=8, N=2048, D=1024, H=4, E=64 on gfx1250 (MI455X).
// Pipeline:
//   0) fp32 -> f16 convert pass (x, Wqk, Wv, Wo) so GEMM loops are pure f16
//   1) fused QK/V projection GEMM (WMMA f16, 32x64 tile/wave), scatter to
//      head-major Q/K and transposed Vt[b,h,e,n]
//   2) flash attention per (b,h,16-query tile): QK^T WMMA, online softmax via
//      16-lane shfl reductions, P re-layout through LDS, PV WMMA
//   3) output projection GEMM (WMMA f16, 32x64 tile/wave) + bias -> fp32 out
// Workspace: xh(32MB) Wqkh(1MB) Wvh(.5MB) Woh(.5MB) Qh/Kh/Vt/Oh(4x8MB) ~ 66MB

typedef _Float16 v16h __attribute__((ext_vector_type(16)));
typedef _Float16 v8h  __attribute__((ext_vector_type(8)));
typedef float    v8f  __attribute__((ext_vector_type(8)));

#define B_  8
#define N_  2048
#define D_  1024
#define H_  4
#define E_  64
#define HD_ 256

__device__ __forceinline__ v8f wmma32(v16h a, v16h b, v8f c) {
  // (neg_a, A, neg_b, B, c_mod, C, reuse_a, reuse_b)
  return __builtin_amdgcn_wmma_f32_16x16x32_f16(false, a, false, b, (short)0, c,
                                                false, false);
}

// A fragment (16x32 f16): lane holds row m = lane&15.
// hlf = lane>>4: elems 0..15 = K kb..kb+15, kb=16*hlf.
__device__ __forceinline__ v16h afrag_f16(const _Float16* __restrict__ rowp, int hlf) {
  const int kb = hlf * 16;
  v8h lo = *(const v8h*)(rowp + kb);
  v8h hi = *(const v8h*)(rowp + kb + 8);
  return __builtin_shufflevector(lo, hi, 0, 1, 2, 3, 4, 5, 6, 7,
                                 8, 9, 10, 11, 12, 13, 14, 15);
}

// B fragment (32x16 f16): lane holds column n = lane&15; elems = K 16*hlf..+15.
__device__ __forceinline__ v16h bfrag_f16(const _Float16* __restrict__ colp, int hlf) {
  return *(const v16h*)(colp + hlf * 16);
}

// ---------------------------------------------------------------------------
// Kernel 0: elementwise fp32 -> f16, 8 elements/thread, vectorized.
// ---------------------------------------------------------------------------
__global__ __launch_bounds__(256) void cvt_f16_kernel(
    const float* __restrict__ src, _Float16* __restrict__ dst, int n8) {
  const int i = blockIdx.x * blockDim.x + threadIdx.x;
  if (i >= n8) return;
  const float4* __restrict__ s = (const float4*)src + (size_t)i * 2;
  const float4 a = s[0];
  const float4 b = s[1];
  v8h h;
  h[0] = (_Float16)a.x; h[1] = (_Float16)a.y; h[2] = (_Float16)a.z; h[3] = (_Float16)a.w;
  h[4] = (_Float16)b.x; h[5] = (_Float16)b.y; h[6] = (_Float16)b.z; h[7] = (_Float16)b.w;
  *((v8h*)dst + i) = h;
}

// ---------------------------------------------------------------------------
// Kernel 1: fused QK/V projection. 32x64 output tile per wave, pure f16.
// Each B (weight) fragment is reused by two A row-tiles: 16 WMMA per K-step-64
// against 4 A-frag + 8 B-frag loads.
// colblk 0..7  -> qk cols [0,512) scattered to Qh/Kh (last-axis split (H,E,2))
// colblk 8..11 -> v  cols [0,256) scattered transposed into Vt[b,h,e,n]
// ---------------------------------------------------------------------------
__global__ __launch_bounds__(256) void qkv_proj_kernel(
    const _Float16* __restrict__ xh, const _Float16* __restrict__ Wqkh,
    const _Float16* __restrict__ Wvh, const float* __restrict__ bqk,
    const float* __restrict__ bv, _Float16* __restrict__ Qh,
    _Float16* __restrict__ Kh, _Float16* __restrict__ Vt) {
  const int wave = threadIdx.x >> 5;
  const int lane = threadIdx.x & 31;
  const int hlf = lane >> 4;
  const int l16 = lane & 15;

  const int job = blockIdx.x * 8 + wave;  // 768 blocks * 8 waves = 6144 jobs
  const int rowtile = job / 12;           // 512 row tiles of 32 (M = 16384)
  const int colblk = job - rowtile * 12;  // 12 column blocks of 64
  const int row0 = rowtile * 32;
  const bool isv = (colblk >= 8);
  const _Float16* __restrict__ W = isv ? Wvh : Wqkh;
  const int cbase = isv ? (colblk - 8) * 64 : colblk * 64;

  v8f acc[2][4] = {};
  const _Float16* __restrict__ xrow0 = xh + (size_t)(row0 + l16) * D_;
  const _Float16* __restrict__ xrow1 = xh + (size_t)(row0 + 16 + l16) * D_;
  const _Float16* __restrict__ w0 = W + (size_t)(cbase + 0 + l16) * D_;
  const _Float16* __restrict__ w1 = W + (size_t)(cbase + 16 + l16) * D_;
  const _Float16* __restrict__ w2 = W + (size_t)(cbase + 32 + l16) * D_;
  const _Float16* __restrict__ w3 = W + (size_t)(cbase + 48 + l16) * D_;

  for (int k0 = 0; k0 < D_; k0 += 64) {
    __builtin_prefetch(xrow0 + k0 + 256, 0, 3);  // global_prefetch_b8
    __builtin_prefetch(xrow1 + k0 + 256, 0, 3);
    // Issue ALL loads of the unrolled step before any WMMA -> loads batch
    // in clauses under partial waits instead of serializing.
    const v16h a00 = afrag_f16(xrow0 + k0, hlf);
    const v16h a01 = afrag_f16(xrow0 + k0 + 32, hlf);
    const v16h a10 = afrag_f16(xrow1 + k0, hlf);
    const v16h a11 = afrag_f16(xrow1 + k0 + 32, hlf);
    const v16h b00 = bfrag_f16(w0 + k0, hlf);
    const v16h b10 = bfrag_f16(w1 + k0, hlf);
    const v16h b20 = bfrag_f16(w2 + k0, hlf);
    const v16h b30 = bfrag_f16(w3 + k0, hlf);
    const v16h b01 = bfrag_f16(w0 + k0 + 32, hlf);
    const v16h b11 = bfrag_f16(w1 + k0 + 32, hlf);
    const v16h b21 = bfrag_f16(w2 + k0 + 32, hlf);
    const v16h b31 = bfrag_f16(w3 + k0 + 32, hlf);
    acc[0][0] = wmma32(a00, b00, acc[0][0]);
    acc[0][1] = wmma32(a00, b10, acc[0][1]);
    acc[0][2] = wmma32(a00, b20, acc[0][2]);
    acc[0][3] = wmma32(a00, b30, acc[0][3]);
    acc[1][0] = wmma32(a10, b00, acc[1][0]);
    acc[1][1] = wmma32(a10, b10, acc[1][1]);
    acc[1][2] = wmma32(a10, b20, acc[1][2]);
    acc[1][3] = wmma32(a10, b30, acc[1][3]);
    acc[0][0] = wmma32(a01, b01, acc[0][0]);
    acc[0][1] = wmma32(a01, b11, acc[0][1]);
    acc[0][2] = wmma32(a01, b21, acc[0][2]);
    acc[0][3] = wmma32(a01, b31, acc[0][3]);
    acc[1][0] = wmma32(a11, b01, acc[1][0]);
    acc[1][1] = wmma32(a11, b11, acc[1][1]);
    acc[1][2] = wmma32(a11, b21, acc[1][2]);
    acc[1][3] = wmma32(a11, b31, acc[1][3]);
  }

#pragma unroll
  for (int rt = 0; rt < 2; ++rt) {
#pragma unroll
    for (int c = 0; c < 4; ++c) {
      const int col = cbase + c * 16 + l16;
#pragma unroll
      for (int r = 0; r < 8; ++r) {
        const int row = row0 + rt * 16 + r + 8 * hlf;  // global (b*N + n)
        const int bb = row >> 11;                      // /N_
        const int nn = row & (N_ - 1);
        float val = acc[rt][c][r];
        if (!isv) {
          val += bqk[col];
          // qk col -> (h, e, s): col = h*128 + e*2 + s
          const int hh = col >> 7;
          const int rem = col & 127;
          const int ee = rem >> 1;
          _Float16* __restrict__ dst = (rem & 1) ? Kh : Qh;
          dst[(((size_t)bb * H_ + hh) * N_ + nn) * E_ + ee] = (_Float16)val;
        } else {
          val += bv[col];
          const int hh = col >> 6;
          const int ee = col & 63;
          // transposed V: Vt[b][h][e][n]
          Vt[(((size_t)bb * H_ + hh) * E_ + ee) * N_ + nn] = (_Float16)val;
        }
      }
    }
  }
}

// ---------------------------------------------------------------------------
// Kernel 2: flash attention. One wave per (b, h, 16-query-row tile).
// 32 keys per step: QK^T (4 WMMA), online softmax (16-lane shfl reductions),
// P re-layout via wave-private LDS slab, PV (4 WMMA into persistent acc).
// Faithful to reference: softmax first, then divide by sqrt(HEAD_DIM)=16.
// ---------------------------------------------------------------------------
__global__ __launch_bounds__(256) void flash_attn_kernel(
    const _Float16* __restrict__ Qh, const _Float16* __restrict__ Kh,
    const _Float16* __restrict__ Vt, _Float16* __restrict__ Oh) {
  __shared__ _Float16 lds[8][16][40];  // per-wave 16x32 P tile, padded rows
  const int wave = threadIdx.x >> 5;
  const int lane = threadIdx.x & 31;
  const int hlf = lane >> 4;
  const int l16 = lane & 15;

  const int job = blockIdx.x * 8 + wave;  // 512 blocks * 8 = 4096 jobs
  const int qtile = job & 127;            // N/16 = 128
  const int bh = job >> 7;                // B*H = 32
  const int bb = bh >> 2;
  const int hh = bh & 3;

  const _Float16* __restrict__ Qb = Qh + (size_t)bh * N_ * E_;
  const _Float16* __restrict__ Kb = Kh + (size_t)bh * N_ * E_;
  const _Float16* __restrict__ Vb = Vt + (size_t)bh * E_ * N_;
  const int qrow0 = qtile * 16;

  const _Float16* __restrict__ qrow = Qb + (size_t)(qrow0 + l16) * E_;
  const v16h aq0 = afrag_f16(qrow, hlf);       // K-dim e = 0..31
  const v16h aq1 = afrag_f16(qrow + 32, hlf);  // K-dim e = 32..63

  float m_i[8], l_i[8];
#pragma unroll
  for (int r = 0; r < 8; ++r) {
    m_i[r] = -1e30f;
    l_i[r] = 0.0f;
  }
  v8f acc[4] = {};  // 16 rows x 64 e-cols (4 tiles of 16)

  for (int j0 = 0; j0 < N_; j0 += 32) {
    // scores: 16 q-rows x 32 keys, two 16x16 C tiles
    const _Float16* __restrict__ krow0 = Kb + (size_t)(j0 + l16) * E_;
    const _Float16* __restrict__ krow1 = Kb + (size_t)(j0 + 16 + l16) * E_;
    const v16h bk00 = bfrag_f16(krow0, hlf);
    const v16h bk01 = bfrag_f16(krow0 + 32, hlf);
    const v16h bk10 = bfrag_f16(krow1, hlf);
    const v16h bk11 = bfrag_f16(krow1 + 32, hlf);
    v8f sc0 = {}, sc1 = {};
    sc0 = wmma32(aq0, bk00, sc0);
    sc0 = wmma32(aq1, bk01, sc0);
    sc1 = wmma32(aq0, bk10, sc1);
    sc1 = wmma32(aq1, bk11, sc1);

#pragma unroll
    for (int r = 0; r < 8; ++r) {
      float v0 = sc0[r], v1 = sc1[r];
      // row max across the 16 lanes of this half-wave (row lives in one VGPR)
      float mx = fmaxf(v0, v1);
      mx = fmaxf(mx, __shfl_xor(mx, 1, 32));
      mx = fmaxf(mx, __shfl_xor(mx, 2, 32));
      mx = fmaxf(mx, __shfl_xor(mx, 4, 32));
      mx = fmaxf(mx, __shfl_xor(mx, 8, 32));
      const float mnew = fmaxf(m_i[r], mx);
      const float e0 = __expf(v0 - mnew);
      const float e1 = __expf(v1 - mnew);
      float s = e0 + e1;
      s += __shfl_xor(s, 1, 32);
      s += __shfl_xor(s, 2, 32);
      s += __shfl_xor(s, 4, 32);
      s += __shfl_xor(s, 8, 32);
      const float scale = __expf(m_i[r] - mnew);
      l_i[r] = l_i[r] * scale + s;
      m_i[r] = mnew;
#pragma unroll
      for (int c = 0; c < 4; ++c) acc[c][r] *= scale;
      // stage P (C-layout -> row-major LDS) for A-fragment reload
      const int mrow = r + 8 * hlf;
      lds[wave][mrow][l16] = (_Float16)e0;
      lds[wave][mrow][16 + l16] = (_Float16)e1;
    }
    // wave-private LDS RAW; compiler inserts s_wait_dscnt
    const v16h pA = afrag_f16(&lds[wave][l16][0], hlf);
    const v16h bv0 = bfrag_f16(Vb + (size_t)(0 + l16) * N_ + j0, hlf);
    const v16h bv1 = bfrag_f16(Vb + (size_t)(16 + l16) * N_ + j0, hlf);
    const v16h bv2 = bfrag_f16(Vb + (size_t)(32 + l16) * N_ + j0, hlf);
    const v16h bv3 = bfrag_f16(Vb + (size_t)(48 + l16) * N_ + j0, hlf);
    acc[0] = wmma32(pA, bv0, acc[0]);
    acc[1] = wmma32(pA, bv1, acc[1]);
    acc[2] = wmma32(pA, bv2, acc[2]);
    acc[3] = wmma32(pA, bv3, acc[3]);
  }

#pragma unroll
  for (int r = 0; r < 8; ++r) {
    const int mrow = r + 8 * hlf;
    const float inv = 1.0f / (l_i[r] * 16.0f);  // softmax norm then /sqrt(256)
    const size_t bn = (size_t)bb * N_ + qrow0 + mrow;
#pragma unroll
    for (int c = 0; c < 4; ++c)
      Oh[bn * HD_ + hh * E_ + c * 16 + l16] = (_Float16)(acc[c][r] * inv);
  }
}

// ---------------------------------------------------------------------------
// Kernel 3: output projection. out[BN,1024] = Oh[BN,256] @ Wo^T + bo (fp32),
// 32x64 tile per wave with B-fragment reuse across the two row tiles.
// ---------------------------------------------------------------------------
__global__ __launch_bounds__(256) void out_proj_kernel(
    const _Float16* __restrict__ Oh, const _Float16* __restrict__ Woh,
    const float* __restrict__ bo, float* __restrict__ out) {
  const int wave = threadIdx.x >> 5;
  const int lane = threadIdx.x & 31;
  const int hlf = lane >> 4;
  const int l16 = lane & 15;

  const int job = blockIdx.x * 8 + wave;  // 1024 blocks * 8 = 8192 jobs
  const int rowtile = job >> 4;           // 512 tiles of 32 rows
  const int colblk = job & 15;            // 16 blocks of 64 cols
  const int row0 = rowtile * 32;
  const int cbase = colblk * 64;

  v8f acc[2][4] = {};
  const _Float16* __restrict__ arow0 = Oh + (size_t)(row0 + l16) * HD_;
  const _Float16* __restrict__ arow1 = Oh + (size_t)(row0 + 16 + l16) * HD_;
  const _Float16* __restrict__ w0 = Woh + (size_t)(cbase + 0 + l16) * HD_;
  const _Float16* __restrict__ w1 = Woh + (size_t)(cbase + 16 + l16) * HD_;
  const _Float16* __restrict__ w2 = Woh + (size_t)(cbase + 32 + l16) * HD_;
  const _Float16* __restrict__ w3 = Woh + (size_t)(cbase + 48 + l16) * HD_;

  for (int k0 = 0; k0 < HD_; k0 += 64) {
    const v16h a00 = afrag_f16(arow0 + k0, hlf);
    const v16h a01 = afrag_f16(arow0 + k0 + 32, hlf);
    const v16h a10 = afrag_f16(arow1 + k0, hlf);
    const v16h a11 = afrag_f16(arow1 + k0 + 32, hlf);
    const v16h b00 = bfrag_f16(w0 + k0, hlf);
    const v16h b10 = bfrag_f16(w1 + k0, hlf);
    const v16h b20 = bfrag_f16(w2 + k0, hlf);
    const v16h b30 = bfrag_f16(w3 + k0, hlf);
    const v16h b01 = bfrag_f16(w0 + k0 + 32, hlf);
    const v16h b11 = bfrag_f16(w1 + k0 + 32, hlf);
    const v16h b21 = bfrag_f16(w2 + k0 + 32, hlf);
    const v16h b31 = bfrag_f16(w3 + k0 + 32, hlf);
    acc[0][0] = wmma32(a00, b00, acc[0][0]);
    acc[0][1] = wmma32(a00, b10, acc[0][1]);
    acc[0][2] = wmma32(a00, b20, acc[0][2]);
    acc[0][3] = wmma32(a00, b30, acc[0][3]);
    acc[1][0] = wmma32(a10, b00, acc[1][0]);
    acc[1][1] = wmma32(a10, b10, acc[1][1]);
    acc[1][2] = wmma32(a10, b20, acc[1][2]);
    acc[1][3] = wmma32(a10, b30, acc[1][3]);
    acc[0][0] = wmma32(a01, b01, acc[0][0]);
    acc[0][1] = wmma32(a01, b11, acc[0][1]);
    acc[0][2] = wmma32(a01, b21, acc[0][2]);
    acc[0][3] = wmma32(a01, b31, acc[0][3]);
    acc[1][0] = wmma32(a11, b01, acc[1][0]);
    acc[1][1] = wmma32(a11, b11, acc[1][1]);
    acc[1][2] = wmma32(a11, b21, acc[1][2]);
    acc[1][3] = wmma32(a11, b31, acc[1][3]);
  }

#pragma unroll
  for (int rt = 0; rt < 2; ++rt) {
#pragma unroll
    for (int c = 0; c < 4; ++c) {
      const int col = cbase + c * 16 + l16;
      const float bias = bo[col];
#pragma unroll
      for (int r = 0; r < 8; ++r) {
        const size_t row = (size_t)row0 + rt * 16 + r + 8 * hlf;
        out[row * D_ + col] = acc[rt][c][r] + bias;
      }
    }
  }
}

extern "C" void kernel_launch(void* const* d_in, const int* in_sizes, int n_in,
                              void* d_out, int out_size, void* d_ws,
                              size_t ws_size, hipStream_t stream) {
  const float* x = (const float*)d_in[0];
  const float* Wqk = (const float*)d_in[1];
  const float* bqk = (const float*)d_in[2];
  const float* Wv = (const float*)d_in[3];
  const float* bv = (const float*)d_in[4];
  const float* Wo = (const float*)d_in[5];
  const float* bo = (const float*)d_in[6];
  float* out = (float*)d_out;

  const size_t nx = (size_t)B_ * N_ * D_;    // 16,777,216
  const size_t nqk = (size_t)2 * HD_ * D_;   // 524,288
  const size_t nv = (size_t)HD_ * D_;        // 262,144
  const size_t nwo = (size_t)D_ * HD_;       // 262,144
  const size_t per = (size_t)B_ * H_ * N_ * E_;  // 4,194,304

  _Float16* xh = (_Float16*)d_ws;
  _Float16* Wqkh = xh + nx;
  _Float16* Wvh = Wqkh + nqk;
  _Float16* Woh = Wvh + nv;
  _Float16* Qh = Woh + nwo;
  _Float16* Kh = Qh + per;
  _Float16* Vt = Kh + per;
  _Float16* Oh = Vt + per;

  // 0) fp32 -> f16 conversion passes (8 elems/thread)
  cvt_f16_kernel<<<(int)(nx / 8 / 256), 256, 0, stream>>>(x, xh, (int)(nx / 8));
  cvt_f16_kernel<<<(int)(nqk / 8 / 256), 256, 0, stream>>>(Wqk, Wqkh, (int)(nqk / 8));
  cvt_f16_kernel<<<(int)(nv / 8 / 256), 256, 0, stream>>>(Wv, Wvh, (int)(nv / 8));
  cvt_f16_kernel<<<(int)(nwo / 8 / 256), 256, 0, stream>>>(Wo, Woh, (int)(nwo / 8));

  // 1) QK/V projection, 2) flash attention, 3) output projection
  qkv_proj_kernel<<<768, 256, 0, stream>>>(xh, Wqkh, Wvh, bqk, bv, Qh, Kh, Vt);
  flash_attn_kernel<<<512, 256, 0, stream>>>(Qh, Kh, Vt, Oh);
  out_proj_kernel<<<1024, 256, 0, stream>>>(Oh, Woh, bo, out);

  (void)in_sizes;
  (void)n_in;
  (void)out_size;
  (void)ws_size;
}
